// Critic_5978594476494
// MI455X (gfx1250) — compile-verified, matrix operand order checked by reading, not run on hardware
//
// CDNA5 / gfx1250 LSTM-with-resets (T=4096, B=32, D=512, H=256) + projection.
//
// Plan (MI455X):
//  k_cvt     : fp32 -> bf16 for x, W_ih, W_hh (parallel, HBM-trivial at 23.3 TB/s)
//  k_gate_gemm: G[t,b,4H] = x @ W_ih^T + b  -- 137 GFLOP bf16 WMMA GEMM,
//               tiles staged to LDS via the Tensor Data Mover (tensor_load_to_lds,
//               double buffered, s_wait_tensorcnt).
//  k_scan    : sequential recurrence. One 4-WGP cluster; each WG holds its
//              4-gate x 64-hidden W_hh slice (128KB bf16) in LDS, does
//              h @ W_hh_slice^T with v_wmma_f32_16x16x32_bf16, exchanges h via a
//              global bf16 hs history buffer with fence + s_cluster_barrier
//              (+ atomic-counter fallback barrier) once per step.
//  k_proj    : hs @ W_proj + b (memory bound, wave-per-row shuffle reduce).
//
// Workspace layout (assumed large enough): G fp32 512MB, hs bf16 64MB,
// x/W bf16 copies ~135MB, barrier counter.

#include <hip/hip_runtime.h>
#include <stdint.h>

#define T_STEPS 4096
#define BATCH   32
#define DIN     512
#define HID     256
#define G4H     1024  // 4*HID

typedef __bf16 bf16;
typedef bf16  v8bf  __attribute__((ext_vector_type(8)));
typedef bf16  v16bf __attribute__((ext_vector_type(16)));
typedef float v8f   __attribute__((ext_vector_type(8)));
typedef unsigned int u32x4 __attribute__((ext_vector_type(4)));
typedef int   i32x4 __attribute__((ext_vector_type(4)));
typedef int   i32x8 __attribute__((ext_vector_type(8)));

#define HAVE_TDM __has_builtin(__builtin_amdgcn_tensor_load_to_lds)

// ---------- small helpers ----------
__device__ __forceinline__ uint16_t f32_to_bf16(float f) {
  uint32_t u = __float_as_uint(f);
  u += 0x7FFFu + ((u >> 16) & 1u);           // round-to-nearest-even
  return (uint16_t)(u >> 16);
}
__device__ __forceinline__ float bf16_to_f32(uint16_t h) {
  return __uint_as_float(((uint32_t)h) << 16);
}
__device__ __forceinline__ float sigf(float x)    { return 1.0f / (1.0f + __expf(-x)); }
__device__ __forceinline__ float tanhfast(float x){ return 2.0f * sigf(2.0f * x) - 1.0f; }

union B16x16 { v16bf v; uint32_t u[8]; };
__device__ __forceinline__ v16bf zero16() {
  B16x16 t;
  #pragma unroll
  for (int i = 0; i < 8; i++) t.u[i] = 0u;
  return t.v;
}
__device__ __forceinline__ v16bf cat16(v8bf a, v8bf b) {
  return __builtin_shufflevector(a, b, 0,1,2,3,4,5,6,7,8,9,10,11,12,13,14,15);
}
// Build a 16-bit A/B WMMA operand: per ISA 7.12.2, lane L owns row/col (L%16)
// and two contiguous 8-element K runs at k0 = 8*(L/16) and k0+16.
__device__ __forceinline__ v16bf load_op16(const uint16_t* p) {
  return cat16(*(const v8bf*)p, *(const v8bf*)(p + 16));
}
__device__ __forceinline__ v8f wmma_bf16(v16bf a, v16bf b, v8f c) {
  return __builtin_amdgcn_wmma_f32_16x16x32_bf16(false, a, false, b, (short)0, c,
                                                 false, false);
}

#if HAVE_TDM
// 2D TDM descriptor: tile [tile_d1 rows][tile_d0 elems] of `elem_code`-sized
// elements (1 => 2 bytes), row stride `stride0` elems, dense into LDS.
__device__ __forceinline__ void tdm_load_2d(uint32_t lds_addr, const void* gptr,
                                            uint32_t tensor_d0, uint32_t tensor_d1,
                                            uint64_t stride0,
                                            uint32_t tile_d0, uint32_t tile_d1,
                                            uint32_t elem_code) {
  uint64_t ga = (uint64_t)gptr;
  u32x4 g0;
  g0[0] = 1u;                                            // count=1, no gather
  g0[1] = lds_addr;                                      // LDS byte address
  g0[2] = (uint32_t)ga;                                  // global_addr[31:0]
  g0[3] = (uint32_t)((ga >> 32) & 0x01FFFFFFu) | 0x80000000u;  // addr[56:32] | type=2
  i32x8 g1;
  g1[0] = (int)(elem_code << 16);                        // wg_mask=0, data_size
  g1[1] = (int)((tensor_d0 & 0xFFFFu) << 16);            // tensor_dim0[15:0]
  g1[2] = (int)((tensor_d0 >> 16) | ((tensor_d1 & 0xFFFFu) << 16));
  g1[3] = (int)((tensor_d1 >> 16) | (tile_d0 << 16));    // tile_dim0
  g1[4] = (int)(tile_d1 & 0xFFFFu);                      // tile_dim1, tile_dim2=0
  g1[5] = (int)(uint32_t)(stride0 & 0xFFFFFFFFu);        // tensor_dim0_stride lo
  g1[6] = (int)(uint32_t)((stride0 >> 32) & 0xFFFFu);    // stride hi, dim1_stride=0
  g1[7] = 0;
  i32x4 z4 = {0, 0, 0, 0};                               // groups 2/3 unused (2D)
  i32x8 z8 = {0, 0, 0, 0, 0, 0, 0, 0};                   // extra group (clang-23 form)
  __builtin_amdgcn_tensor_load_to_lds(g0, g1, z4, z4, z8, 0);
}
#endif

// ---------- kernel 0: fp32 -> bf16 ----------
__global__ void k_cvt(const float* __restrict__ src, uint16_t* __restrict__ dst,
                      size_t n) {
  size_t i = (size_t)blockIdx.x * blockDim.x + threadIdx.x;
  size_t stride = (size_t)gridDim.x * blockDim.x;
  for (; i < n; i += stride) dst[i] = f32_to_bf16(src[i]);
}

__global__ void k_init(unsigned* cnt) {
  if (threadIdx.x == 0 && blockIdx.x == 0) *cnt = 0u;
}

// ---------- kernel 1: G = x @ W_ih^T + b_ih + b_hh ----------
// Block = 256 threads = 8 waves, tile 128(M) x 128(N), K chunks of 32.
// Wave (wm 0..3, wn 0..1) owns 32(M) x 64(N) -> 2x4 accumulator tiles.
__global__ __launch_bounds__(256) void k_gate_gemm(
    const uint16_t* __restrict__ xb,   // [T*B][512] bf16
    const uint16_t* __restrict__ wb,   // [1024][512] bf16
    const float* __restrict__ b_ih, const float* __restrict__ b_hh,
    float* __restrict__ G) {           // [T*B][1024] fp32
  __shared__ uint16_t shA[2][128][32];
  __shared__ uint16_t shB[2][128][32];
  const int tid = threadIdx.x, lane = tid & 31, wv = tid >> 5;
  const int wm = wv & 3, wn = wv >> 2;
  const int col16 = lane & 15, khalf = lane >> 4;
  const int mBase = blockIdx.y * 128;
  const int nBase = blockIdx.x * 128;

  v8f acc[2][4];
  #pragma unroll
  for (int i = 0; i < 2; i++)
    #pragma unroll
    for (int j = 0; j < 4; j++) acc[i][j] = {};

#if HAVE_TDM
  if (wv == 0) {  // one DMA per descriptor; TDM ignores EXEC
    tdm_load_2d((uint32_t)(uintptr_t)&shA[0][0][0], xb + (size_t)mBase * DIN,
                DIN, (uint32_t)(T_STEPS * BATCH), DIN, 32, 128, 1);
    tdm_load_2d((uint32_t)(uintptr_t)&shB[0][0][0], wb + (size_t)nBase * DIN,
                DIN, G4H, DIN, 32, 128, 1);
  }
#endif

  for (int kk = 0; kk < 16; kk++) {
    const int buf = kk & 1;
    __syncthreads();  // buffer (kk+1)&1 is free again
#if HAVE_TDM
    if (wv == 0) {
      if (kk + 1 < 16) {
        tdm_load_2d((uint32_t)(uintptr_t)&shA[buf ^ 1][0][0],
                    xb + ((size_t)mBase * DIN + (kk + 1) * 32),
                    DIN, (uint32_t)(T_STEPS * BATCH), DIN, 32, 128, 1);
        tdm_load_2d((uint32_t)(uintptr_t)&shB[buf ^ 1][0][0],
                    wb + ((size_t)nBase * DIN + (kk + 1) * 32),
                    DIN, G4H, DIN, 32, 128, 1);
        __builtin_amdgcn_s_wait_tensorcnt(2);  // stage kk done (in-order)
      } else {
        __builtin_amdgcn_s_wait_tensorcnt(0);  // drain
      }
    }
#else
    {  // fallback staging: 256 threads copy 128x32 bf16 tiles
      const int r = tid >> 1, o = (tid & 1) * 16;
      const uint16_t* sa = xb + (size_t)(mBase + r) * DIN + kk * 32 + o;
      const uint16_t* sb = wb + (size_t)(nBase + r) * DIN + kk * 32 + o;
      *(v8bf*)&shA[buf][r][o]     = *(const v8bf*)sa;
      *(v8bf*)&shA[buf][r][o + 8] = *(const v8bf*)(sa + 8);
      *(v8bf*)&shB[buf][r][o]     = *(const v8bf*)sb;
      *(v8bf*)&shB[buf][r][o + 8] = *(const v8bf*)(sb + 8);
    }
#endif
    __syncthreads();  // tile kk visible to all waves

    v16bf A[2];
    #pragma unroll
    for (int mt = 0; mt < 2; mt++)
      A[mt] = load_op16(&shA[buf][wm * 32 + mt * 16 + col16][khalf * 8]);
    #pragma unroll
    for (int nt = 0; nt < 4; nt++) {
      v16bf Bv = load_op16(&shB[buf][wn * 64 + nt * 16 + col16][khalf * 8]);
      acc[0][nt] = wmma_bf16(A[0], Bv, acc[0][nt]);
      acc[1][nt] = wmma_bf16(A[1], Bv, acc[1][nt]);
    }
  }

  // D layout: lane L -> n = L%16; vgpr r -> m = r + 8*(L/16)
  #pragma unroll
  for (int nt = 0; nt < 4; nt++) {
    const int n = nBase + wn * 64 + nt * 16 + col16;
    const float bias = b_ih[n] + b_hh[n];
    #pragma unroll
    for (int mt = 0; mt < 2; mt++)
      #pragma unroll
      for (int r = 0; r < 8; r++) {
        const int m = mBase + wm * 32 + mt * 16 + r + 8 * khalf;
        G[(size_t)m * G4H + n] = acc[mt][nt][r] + bias;
      }
  }
}

// ---------- kernel 2: sequential LSTM scan, 4-WGP cluster ----------
// WG w owns hidden slice [64w, 64w+64). Wave (mt = wv>>2, jb = wv&3) owns
// batch rows [16mt,16mt+16) x hidden block [16jb,16jb+16) and ALL four gates
// of that block (4 accumulator tiles) so the c/h update is pure per-lane VALU.
__global__ __launch_bounds__(256) void k_scan(
    const float* __restrict__ G,        // [T*B][1024] fp32 preactivations
    const int* __restrict__ reset,      // [T][B]
    const uint16_t* __restrict__ whhb,  // [1024][256] bf16
    uint16_t* __restrict__ hs,          // [(T+1)*B][256] bf16 (slot 0 = zeros)
    unsigned* __restrict__ bar_cnt) {
  extern __shared__ uint16_t Wl[];  // [256][256] bf16 = 128KB (of 320KB LDS)
  const int tid = threadIdx.x, lane = tid & 31, wv = tid >> 5;
  const int mt = wv >> 2, jb = wv & 3;
  const int col16 = lane & 15, khalf = lane >> 4;
  const int wg = blockIdx.x;                 // 0..3
  const unsigned nwg = gridDim.x;            // 4

  // Preload this WG's W_hh slice. Local col n = gate*64 + j' maps to global
  // gate row gate*256 + wg*64 + j'; stored K-contiguous for b128 operand reads.
  {
    const int n = tid;  // 256 threads, 256 rows
    const int gg = n >> 6, jl = n & 63;
    const uint16_t* src = whhb + (size_t)(gg * HID + wg * 64 + jl) * HID;
    uint16_t* dst = &Wl[(size_t)n * HID];
    #pragma unroll 4
    for (int q = 0; q < 32; q++) ((v8bf*)dst)[q] = ((const v8bf*)src)[q];
  }
  // Zero hs slot 0 (this WG's 64 columns of all 32 batch rows).
  {
    const int b = tid >> 3, o = (tid & 7) * 8;
    B16x16 z;
    #pragma unroll
    for (int i = 0; i < 4; i++) z.u[i] = 0u;
    *(v8bf*)(hs + (size_t)b * HID + wg * 64 + o) = *(v8bf*)&z;
  }

  auto step_barrier = [&](unsigned target) {
    __builtin_amdgcn_fence(__ATOMIC_RELEASE, "agent");
    __syncthreads();
    if (tid == 0) {
      __hip_atomic_fetch_add(bar_cnt, 1u, __ATOMIC_RELEASE, __HIP_MEMORY_SCOPE_AGENT);
      while (__hip_atomic_load(bar_cnt, __ATOMIC_ACQUIRE, __HIP_MEMORY_SCOPE_AGENT) < target)
        __builtin_amdgcn_s_sleep(1);
    }
    // Real cluster split-barrier when launched as a 4-WG cluster (NOP otherwise;
    // the atomic counter above provides the device-scope ordering regardless).
    __builtin_amdgcn_s_cluster_barrier();
    __syncthreads();
    __builtin_amdgcn_fence(__ATOMIC_ACQUIRE, "agent");
  };
  step_barrier(nwg);  // weights + hs[0] published

  v8f c = {};  // cell state, D-layout: r -> b = 16mt + 8*khalf + r, j fixed per lane

  for (int t = 0; t < T_STEPS; t++) {
    const int bA = mt * 16 + col16;            // A-operand batch row of this lane
    const int rstA = reset[t * BATCH + bA];

    // Accumulators start at the precomputed input preactivations.
    v8f acc[4];
    #pragma unroll
    for (int gg = 0; gg < 4; gg++) {
      v8f a = {};
      #pragma unroll
      for (int r = 0; r < 8; r++) {
        const int b = mt * 16 + 8 * khalf + r;
        a[r] = G[((size_t)t * BATCH + b) * G4H + gg * HID + wg * 64 + jb * 16 + col16];
      }
      acc[gg] = a;
    }
    if (t + 1 < T_STEPS)  // warm L2/WGP$ for next step's preactivations
      __builtin_prefetch(&G[((size_t)(t + 1) * BATCH) * G4H], 0, 1);

    // gates += h_{t-1} @ W_hh_slice^T  (h zeroed on reset via A operand)
    const uint16_t* hrow = hs + ((size_t)t * BATCH + bA) * HID;
    #pragma unroll
    for (int kk = 0; kk < 8; kk++) {
      v16bf A = load_op16(hrow + kk * 32 + khalf * 8);
      A = rstA ? zero16() : A;
      #pragma unroll
      for (int gg = 0; gg < 4; gg++) {
        v16bf Bv = load_op16(&Wl[(size_t)(gg * 64 + jb * 16 + col16) * HID +
                                 kk * 32 + khalf * 8]);
        acc[gg] = wmma_bf16(A, Bv, acc[gg]);
      }
    }

    // Elementwise LSTM update; store h slice (bf16) into hs slot t+1.
    #pragma unroll
    for (int r = 0; r < 8; r++) {
      const int b = mt * 16 + 8 * khalf + r;
      const float keep = reset[t * BATCH + b] ? 0.0f : 1.0f;
      const float iv = sigf(acc[0][r]);
      const float fv = sigf(acc[1][r]);
      const float gv = tanhfast(acc[2][r]);
      const float ov = sigf(acc[3][r]);
      const float cn = fv * (c[r] * keep) + iv * gv;
      c[r] = cn;
      const float h = ov * tanhfast(cn);
      hs[((size_t)(t + 1) * BATCH + b) * HID + wg * 64 + jb * 16 + col16] =
          f32_to_bf16(h);
    }
    step_barrier(nwg * (unsigned)(t + 2));
  }
}

// ---------- kernel 3: out = hs @ W_proj^T + b_proj ----------
__global__ __launch_bounds__(256) void k_proj(const uint16_t* __restrict__ hs,
                                              const float* __restrict__ wproj,
                                              const float* __restrict__ bproj,
                                              float* __restrict__ out) {
  const int lane = threadIdx.x & 31, wv = threadIdx.x >> 5;
  const size_t row = (size_t)blockIdx.x * 8 + wv;   // row = t*B + b
  if (row >= (size_t)T_STEPS * BATCH) return;
  const uint16_t* p = hs + (row + BATCH) * HID + lane * 8;  // slot t+1
  float s = 0.f;
  #pragma unroll
  for (int i = 0; i < 8; i++) s += bf16_to_f32(p[i]) * wproj[lane * 8 + i];
  #pragma unroll
  for (int m = 16; m >= 1; m >>= 1) s += __shfl_xor(s, m, 32);
  if (lane == 0) out[row] = s + bproj[0];
}

// ---------- host launcher ----------
extern "C" void kernel_launch(void* const* d_in, const int* in_sizes, int n_in,
                              void* d_out, int out_size, void* d_ws, size_t ws_size,
                              hipStream_t stream) {
  const float* x      = (const float*)d_in[0];
  const int*   reset  = (const int*)d_in[1];
  const float* W_ih   = (const float*)d_in[2];
  const float* W_hh   = (const float*)d_in[3];
  const float* b_ih   = (const float*)d_in[4];
  const float* b_hh   = (const float*)d_in[5];
  const float* W_proj = (const float*)d_in[6];
  const float* b_proj = (const float*)d_in[7];
  float* out = (float*)d_out;

  char* ws = (char*)d_ws;
  size_t off = 0;
  auto carve = [&](size_t bytes) {
    void* p = ws + off;
    off = (off + bytes + 255) & ~(size_t)255;
    return p;
  };
  float*    G    = (float*)   carve((size_t)T_STEPS * BATCH * G4H * sizeof(float));
  uint16_t* hs   = (uint16_t*)carve((size_t)(T_STEPS + 1) * BATCH * HID * 2);
  uint16_t* xb   = (uint16_t*)carve((size_t)T_STEPS * BATCH * DIN * 2);
  uint16_t* wihb = (uint16_t*)carve((size_t)G4H * DIN * 2);
  uint16_t* whhb = (uint16_t*)carve((size_t)G4H * HID * 2);
  unsigned* cnt  = (unsigned*)carve(256);

  k_init<<<1, 64, 0, stream>>>(cnt);
  k_cvt<<<2048, 256, 0, stream>>>(x, xb, (size_t)T_STEPS * BATCH * DIN);
  k_cvt<<<512, 256, 0, stream>>>(W_ih, wihb, (size_t)G4H * DIN);
  k_cvt<<<256, 256, 0, stream>>>(W_hh, whhb, (size_t)G4H * HID);

  dim3 g1(G4H / 128, (T_STEPS * BATCH) / 128);  // (8, 1024)
  k_gate_gemm<<<g1, 256, 0, stream>>>(xb, wihb, b_ih, b_hh, G);

  k_scan<<<4, 256, 256 * 256 * 2, stream>>>(G, reset, whhb, hs, cnt);

  k_proj<<<(T_STEPS * BATCH) / 8, 256, 0, stream>>>(hs, W_proj, b_proj, out);
}
